// BaselineCoarseGraphModel_22187801051308
// MI455X (gfx1250) — matrix operand order repaired and checked by reading, not compiled, not run on hardware
//
#include <hip/hip_runtime.h>
#include <hip/hip_bf16.h>

// ---------------------------------------------------------------------------
// Problem constants (from reference)
// ---------------------------------------------------------------------------
constexpr int N_NODES = 12000;
constexpr int IN_DIM  = 256;
constexpr int HID     = 256;
constexpr int KNN_K   = 16;

typedef unsigned short u16;      // raw bf16 bits
typedef __attribute__((ext_vector_type(16))) __bf16 v16bf;
typedef __attribute__((ext_vector_type(8)))  float  v8f;

union BF16Frag { v16bf v; uint4 u[2]; };

__device__ __forceinline__ u16 f32_to_bf16_bits(float f) {
    unsigned int u = __float_as_uint(f);
    // round to nearest even
    unsigned int r = u + 0x7FFFu + ((u >> 16) & 1u);
    return (u16)(r >> 16);
}

__device__ __forceinline__ float gelu_f(float x) {
    return 0.5f * x * (1.0f + erff(x * 0.70710678118654752440f));
}

// ---------------------------------------------------------------------------
// f32 -> bf16 bits elementwise conversion
// ---------------------------------------------------------------------------
__global__ void __launch_bounds__(256)
cvt_f32_bf16_kernel(const float* __restrict__ in, u16* __restrict__ out, int n) {
    int i = blockIdx.x * 256 + threadIdx.x;
    if (i < n) out[i] = f32_to_bf16_bits(in[i]);
}

// ---------------------------------------------------------------------------
// KNN: for each node i, find 16 nearest (excluding self) among 12000 2-D
// points; emit indices and normalized inverse-distance weights.
// One thread per node; 512-point tiles staged in LDS; per-thread top-16
// insertion lists kept in LDS (stride-256 layout -> conflict free).
// ---------------------------------------------------------------------------
constexpr int KNN_TILE = 512;

__global__ void __launch_bounds__(256)
knn_kernel(const float* __restrict__ cents, int* __restrict__ idxOut,
           float* __restrict__ wOut, int n) {
    __shared__ float2 tile[KNN_TILE];
    __shared__ float  sd[KNN_K * 256];
    __shared__ int    si[KNN_K * 256];

    const int tid = threadIdx.x;
    const int i   = blockIdx.x * 256 + tid;
    const bool active = (i < n);

    float xi = 0.0f, yi = 0.0f;
    if (active) {
        float2 c = reinterpret_cast<const float2*>(cents)[i];
        xi = c.x; yi = c.y;
    }
    #pragma unroll
    for (int t = 0; t < KNN_K; ++t) {
        sd[t * 256 + tid] = 3.0e38f;
        si[t * 256 + tid] = 0;
    }
    float worst = 3.0e38f;

    for (int jb = 0; jb < n; jb += KNN_TILE) {
        int cnt = min(KNN_TILE, n - jb);
        __syncthreads();
        for (int t = tid; t < cnt; t += 256)
            tile[t] = reinterpret_cast<const float2*>(cents)[jb + t];
        __syncthreads();
        if (active) {
            for (int t = 0; t < cnt; ++t) {
                float dx = tile[t].x - xi;
                float dy = tile[t].y - yi;
                float d2 = dx * dx + dy * dy;
                int   j  = jb + t;
                if (d2 < worst && j != i) {
                    int p = KNN_K - 1;
                    while (p > 0 && sd[(p - 1) * 256 + tid] > d2) {
                        sd[p * 256 + tid] = sd[(p - 1) * 256 + tid];
                        si[p * 256 + tid] = si[(p - 1) * 256 + tid];
                        --p;
                    }
                    sd[p * 256 + tid] = d2;
                    si[p * 256 + tid] = j;
                    worst = sd[(KNN_K - 1) * 256 + tid];
                }
            }
        }
    }

    if (active) {
        float inv[KNN_K];
        float s = 0.0f;
        #pragma unroll
        for (int t = 0; t < KNN_K; ++t) {
            float d = sqrtf(fmaxf(sd[t * 256 + tid], 0.0f));
            float v = 1.0f / fmaxf(d, 1e-4f);
            inv[t] = v; s += v;
        }
        float sInv = 1.0f / fmaxf(s, 1e-8f);
        #pragma unroll
        for (int t = 0; t < KNN_K; ++t) {
            idxOut[i * KNN_K + t] = si[t * 256 + tid];
            wOut[i * KNN_K + t]   = inv[t] * sInv;
        }
    }
}

// ---------------------------------------------------------------------------
// Weighted neighbor aggregation: agg[n,:] = sum_k w[n,k] * h[idx[n,k],:]
// 64 threads per node (float4 per thread), 4 nodes per 256-thread block.
// Also emits a bf16 shadow copy for the downstream WMMA GEMM.
// ---------------------------------------------------------------------------
__global__ void __launch_bounds__(256)
aggregate_kernel(const float* __restrict__ h, const int* __restrict__ idx,
                 const float* __restrict__ w, float* __restrict__ agg,
                 u16* __restrict__ aggB, int n) {
    const int tid = threadIdx.x;
    const int nl  = tid >> 6;          // 0..3 node within block
    const int cq  = tid & 63;          // column quad 0..63
    const int node = blockIdx.x * 4 + nl;
    if (node >= n) return;

    const int*   ip = idx + node * KNN_K;
    const float* wp = w   + node * KNN_K;
    float4 acc = make_float4(0.f, 0.f, 0.f, 0.f);
    #pragma unroll
    for (int k = 0; k < KNN_K; ++k) {
        int   j  = ip[k];
        float wk = wp[k];
        float4 hv = *reinterpret_cast<const float4*>(h + (size_t)j * HID + cq * 4);
        acc.x += wk * hv.x; acc.y += wk * hv.y;
        acc.z += wk * hv.z; acc.w += wk * hv.w;
    }
    *reinterpret_cast<float4*>(agg + (size_t)node * HID + cq * 4) = acc;

    union { u16 s[4]; uint2 v; } ub;
    ub.s[0] = f32_to_bf16_bits(acc.x); ub.s[1] = f32_to_bf16_bits(acc.y);
    ub.s[2] = f32_to_bf16_bits(acc.z); ub.s[3] = f32_to_bf16_bits(acc.w);
    *reinterpret_cast<uint2*>(aggB + (size_t)node * HID + cq * 4) = ub.v;
}

// ---------------------------------------------------------------------------
// WMMA GEMM fragment loaders
//
// A fragment (16x32 bf16), lane = (aRow = lane&15, half = lane>>4):
//   VGPR0-3 : K = k0 + half*8 + 0..7        (contiguous 16B load)
//   VGPR4-7 : K = k0 + 16 + half*8 + 0..7   (contiguous 16B load)
// B fragment (32x16 bf16), lane = (col n = lane&15, half = lane>>4):
//   VGPR0-3 : K = k0 + half*8 + 0..7        (contiguous 16B load)
//   VGPR4-7 : K = k0 + 16 + half*8 + 0..7   (contiguous 16B load)
// ---------------------------------------------------------------------------
__device__ __forceinline__ void load_a_frag(BF16Frag& f, const u16* __restrict__ aBase,
                                            int k0, int half) {
    f.u[0] = *reinterpret_cast<const uint4*>(aBase + k0 + half * 8);
    f.u[1] = *reinterpret_cast<const uint4*>(aBase + k0 + 16 + half * 8);
}

__device__ __forceinline__ void load_b_frags(BF16Frag* f, const u16* __restrict__ W,
                                             int wRowBase, int inDim, int k0, int half) {
    #pragma unroll
    for (int t = 0; t < 4; ++t) {
        const u16* bp = W + (size_t)(wRowBase + t * 16) * inDim + k0 + half * 8;
        f[t].u[0] = *reinterpret_cast<const uint4*>(bp);
        f[t].u[1] = *reinterpret_cast<const uint4*>(bp + 16);
    }
}

// ---------------------------------------------------------------------------
// WMMA GEMM:  outF[N,OUT] = act( A[N,IN] @ W[OUT,IN]^T  (+ A2 @ W2^T) + bias )
// bf16 inputs, f32 accumulate via v_wmma_f32_16x16x32_bf16.
//
// 256 threads = 8 waves. G = OUT/64 column groups; each wave owns a
// 16-row x 64-col strip (4 accumulator tiles).
//
// K-loop is software-pipelined (ping-pong X/Y fragment sets, 2 K-steps per
// iteration) so prefetch loads for the next K-step stay in flight while the
// current WMMAs execute -> partial s_wait_loadcnt instead of full drains.
// Requires inDim % 64 == 0 (holds: 256).
// ---------------------------------------------------------------------------
__global__ void __launch_bounds__(256)
wmma_gemm_kernel(const u16* __restrict__ A,  const u16* __restrict__ W,
                 const u16* __restrict__ A2, const u16* __restrict__ W2,
                 const float* __restrict__ bias,
                 float* __restrict__ outF, u16* __restrict__ outB,
                 int n, int inDim, int outDim, int act) {
    const int tid  = threadIdx.x;
    const int lane = tid & 31;
    const int wave = tid >> 5;
    const int G    = outDim >> 6;                 // column groups of 64
    const int colGroup = wave & (G - 1);
    const int rowSub   = wave / G;
    const int rowTilesPerBlock = 8 / G;
    const int m0 = (blockIdx.x * rowTilesPerBlock + rowSub) * 16;
    if (m0 >= n) return;                          // wave-uniform exit

    const int aRow = lane & 15;
    const int half = lane >> 4;
    const int colBase  = colGroup * 64;
    const int wRowBase = colBase + aRow;          // W row for this lane's column

    v8f acc[4];
    #pragma unroll
    for (int t = 0; t < 4; ++t) acc[t] = (v8f){0.f,0.f,0.f,0.f,0.f,0.f,0.f,0.f};

    BF16Frag aX, aY, bX[4], bY[4];

    // ---- pair 1: A @ W^T, pipelined over K ----
    {
        const u16* aBase = A + (size_t)(m0 + aRow) * inDim;
        load_a_frag(aX, aBase, 0, half);
        load_b_frags(bX, W, wRowBase, inDim, 0, half);
        for (int k0 = 0; k0 < inDim; k0 += 64) {
            load_a_frag(aY, aBase, k0 + 32, half);
            load_b_frags(bY, W, wRowBase, inDim, k0 + 32, half);
            #pragma unroll
            for (int t = 0; t < 4; ++t)
                acc[t] = __builtin_amdgcn_wmma_f32_16x16x32_bf16(
                    false, aX.v, false, bX[t].v, (short)0, acc[t], false, false);
            if (k0 + 64 < inDim) {
                load_a_frag(aX, aBase, k0 + 64, half);
                load_b_frags(bX, W, wRowBase, inDim, k0 + 64, half);
            }
            #pragma unroll
            for (int t = 0; t < 4; ++t)
                acc[t] = __builtin_amdgcn_wmma_f32_16x16x32_bf16(
                    false, aY.v, false, bY[t].v, (short)0, acc[t], false, false);
        }
    }

    // ---- optional fused pair 2 (SAGE: + agg @ wn^T), same pipeline ----
    if (A2 != nullptr) {
        const u16* aBase = A2 + (size_t)(m0 + aRow) * inDim;
        load_a_frag(aX, aBase, 0, half);
        load_b_frags(bX, W2, wRowBase, inDim, 0, half);
        for (int k0 = 0; k0 < inDim; k0 += 64) {
            load_a_frag(aY, aBase, k0 + 32, half);
            load_b_frags(bY, W2, wRowBase, inDim, k0 + 32, half);
            #pragma unroll
            for (int t = 0; t < 4; ++t)
                acc[t] = __builtin_amdgcn_wmma_f32_16x16x32_bf16(
                    false, aX.v, false, bX[t].v, (short)0, acc[t], false, false);
            if (k0 + 64 < inDim) {
                load_a_frag(aX, aBase, k0 + 64, half);
                load_b_frags(bX, W2, wRowBase, inDim, k0 + 64, half);
            }
            #pragma unroll
            for (int t = 0; t < 4; ++t)
                acc[t] = __builtin_amdgcn_wmma_f32_16x16x32_bf16(
                    false, aY.v, false, bY[t].v, (short)0, acc[t], false, false);
        }
    }

    // ---- epilogue: bias + activation + stores (f32, optional bf16 shadow) ----
    #pragma unroll
    for (int t = 0; t < 4; ++t) {
        int col = colBase + t * 16 + aRow;
        float bv = (bias != nullptr) ? bias[col] : 0.0f;
        #pragma unroll
        for (int r = 0; r < 8; ++r) {
            int row = m0 + half * 8 + r;
            float v = acc[t][r] + bv;
            if (act) v = gelu_f(v);
            outF[(size_t)row * outDim + col] = v;
            if (outB != nullptr)
                outB[(size_t)row * outDim + col] = f32_to_bf16_bits(v);
        }
    }
}

// ---------------------------------------------------------------------------
// LayerNorm + residual:  h[n,:] += LN(tmp[n,:]) * g + b ;  bf16 shadow copy.
// One wave32 per row, shfl_xor reductions over 256 columns.
// ---------------------------------------------------------------------------
__global__ void __launch_bounds__(256)
ln_residual_kernel(const float* __restrict__ tmp, const float* __restrict__ g,
                   const float* __restrict__ b, float* __restrict__ h,
                   u16* __restrict__ hB, int n) {
    const int lane = threadIdx.x & 31;
    const int wv   = threadIdx.x >> 5;
    const int row  = blockIdx.x * 8 + wv;
    if (row >= n) return;

    const float* tr = tmp + (size_t)row * HID;
    float x[8], s = 0.0f, s2 = 0.0f;
    #pragma unroll
    for (int j = 0; j < 8; ++j) {
        x[j] = tr[lane + 32 * j];
        s  += x[j];
        s2 += x[j] * x[j];
    }
    #pragma unroll
    for (int m = 16; m >= 1; m >>= 1) {
        s  += __shfl_xor(s,  m, 32);
        s2 += __shfl_xor(s2, m, 32);
    }
    const float mu  = s * (1.0f / HID);
    const float var = fmaxf(s2 * (1.0f / HID) - mu * mu, 0.0f);
    const float rs  = rsqrtf(var + 1e-5f);

    float* hr = h + (size_t)row * HID;
    #pragma unroll
    for (int j = 0; j < 8; ++j) {
        int c = lane + 32 * j;
        float y   = (x[j] - mu) * rs * g[c] + b[c];
        float out = hr[c] + y;
        hr[c] = out;
        hB[(size_t)row * HID + c] = f32_to_bf16_bits(out);
    }
}

// ---------------------------------------------------------------------------
// Final head:  out[n] = sigmoid( z1[n,:] . w2 + b2 )   (128-dim dot, f32)
// ---------------------------------------------------------------------------
__global__ void __launch_bounds__(256)
cls_final_kernel(const float* __restrict__ z1, const float* __restrict__ w2,
                 const float* __restrict__ b2, float* __restrict__ out, int n) {
    const int lane = threadIdx.x & 31;
    const int wv   = threadIdx.x >> 5;
    const int row  = blockIdx.x * 8 + wv;
    if (row >= n) return;

    const float* zr = z1 + (size_t)row * 128;
    float s = 0.0f;
    #pragma unroll
    for (int j = 0; j < 4; ++j) s += zr[lane + 32 * j] * w2[lane + 32 * j];
    #pragma unroll
    for (int m = 16; m >= 1; m >>= 1) s += __shfl_xor(s, m, 32);
    if (lane == 0) out[row] = 1.0f / (1.0f + expf(-(s + b2[0])));
}

// ---------------------------------------------------------------------------
// Orchestration
// ---------------------------------------------------------------------------
extern "C" void kernel_launch(void* const* d_in, const int* in_sizes, int n_in,
                              void* d_out, int out_size, void* d_ws, size_t ws_size,
                              hipStream_t stream) {
    (void)in_sizes; (void)n_in; (void)out_size; (void)ws_size;

    const float* feats  = (const float*)d_in[0];
    const float* cents  = (const float*)d_in[1];
    const float* enc_w  = (const float*)d_in[2];
    const float* enc_b  = (const float*)d_in[3];
    const float* g1_ws  = (const float*)d_in[4];
    const float* g1_wn  = (const float*)d_in[5];
    const float* g1_g   = (const float*)d_in[6];
    const float* g1_b   = (const float*)d_in[7];
    const float* g2_ws  = (const float*)d_in[8];
    const float* g2_wn  = (const float*)d_in[9];
    const float* g2_g   = (const float*)d_in[10];
    const float* g2_b   = (const float*)d_in[11];
    const float* cls_w1 = (const float*)d_in[12];
    const float* cls_b1 = (const float*)d_in[13];
    const float* cls_w2 = (const float*)d_in[14];
    const float* cls_b2 = (const float*)d_in[15];
    float* out = (float*)d_out;

    // --- workspace carve-up (256B aligned) ---
    char* ws = (char*)d_ws;
    size_t off = 0;
    auto carve = [&](size_t bytes) -> char* {
        char* p = ws + off;
        off = (off + bytes + 255) & ~(size_t)255;
        return p;
    };
    const size_t NH   = (size_t)N_NODES * HID;
    int*   idxB   = (int*)  carve((size_t)N_NODES * KNN_K * 4);
    float* wgtB   = (float*)carve((size_t)N_NODES * KNN_K * 4);
    float* hF     = (float*)carve(NH * 4);
    u16*   hBf    = (u16*)  carve(NH * 2);
    float* aggF   = (float*)carve(NH * 4);
    u16*   aggBf  = (u16*)  carve(NH * 2);
    float* tmpF   = (float*)carve(NH * 4);
    float* z1F    = (float*)carve((size_t)N_NODES * 128 * 4);
    u16*   featsB = (u16*)  carve(NH * 2);
    u16*   encWB  = (u16*)  carve((size_t)HID * IN_DIM * 2);
    u16*   g1sB   = (u16*)  carve((size_t)HID * HID * 2);
    u16*   g1nB   = (u16*)  carve((size_t)HID * HID * 2);
    u16*   g2sB   = (u16*)  carve((size_t)HID * HID * 2);
    u16*   g2nB   = (u16*)  carve((size_t)HID * HID * 2);
    u16*   c1B    = (u16*)  carve((size_t)128 * HID * 2);

    // 1) KNN graph
    knn_kernel<<<(N_NODES + 255) / 256, 256, 0, stream>>>(cents, idxB, wgtB, N_NODES);

    // 2) bf16 conversions (inputs + weights)
    auto cvt = [&](const float* src, u16* dst, int cnt) {
        cvt_f32_bf16_kernel<<<(cnt + 255) / 256, 256, 0, stream>>>(src, dst, cnt);
    };
    cvt(feats,  featsB, N_NODES * IN_DIM);
    cvt(enc_w,  encWB,  HID * IN_DIM);
    cvt(g1_ws,  g1sB,   HID * HID);
    cvt(g1_wn,  g1nB,   HID * HID);
    cvt(g2_ws,  g2sB,   HID * HID);
    cvt(g2_wn,  g2nB,   HID * HID);
    cvt(cls_w1, c1B,    128 * HID);

    // grid helper: OUT=256 -> 32 rows/block; OUT=128 -> 64 rows/block
    const int grid256 = (N_NODES + 31) / 32;   // 375
    const int grid128 = (N_NODES + 63) / 64;   // 188

    // 3) encoder: h = gelu(feats @ enc_w^T + enc_b), emit f32 + bf16
    wmma_gemm_kernel<<<grid256, 256, 0, stream>>>(
        featsB, encWB, nullptr, nullptr, enc_b, hF, hBf,
        N_NODES, IN_DIM, HID, /*act=*/1);

    // 4-6) SAGE layer 1
    aggregate_kernel<<<(N_NODES + 3) / 4, 256, 0, stream>>>(hF, idxB, wgtB, aggF, aggBf, N_NODES);
    wmma_gemm_kernel<<<grid256, 256, 0, stream>>>(
        hBf, g1sB, aggBf, g1nB, nullptr, tmpF, nullptr,
        N_NODES, HID, HID, /*act=*/1);
    ln_residual_kernel<<<(N_NODES + 7) / 8, 256, 0, stream>>>(tmpF, g1_g, g1_b, hF, hBf, N_NODES);

    // 7-9) SAGE layer 2
    aggregate_kernel<<<(N_NODES + 3) / 4, 256, 0, stream>>>(hF, idxB, wgtB, aggF, aggBf, N_NODES);
    wmma_gemm_kernel<<<grid256, 256, 0, stream>>>(
        hBf, g2sB, aggBf, g2nB, nullptr, tmpF, nullptr,
        N_NODES, HID, HID, /*act=*/1);
    ln_residual_kernel<<<(N_NODES + 7) / 8, 256, 0, stream>>>(tmpF, g2_g, g2_b, hF, hBf, N_NODES);

    // 10) classifier hidden: z1 = gelu(h @ cls_w1^T + cls_b1)  (OUT=128)
    wmma_gemm_kernel<<<grid128, 256, 0, stream>>>(
        hBf, c1B, nullptr, nullptr, cls_b1, z1F, nullptr,
        N_NODES, HID, 128, /*act=*/1);

    // 11) head + sigmoid
    cls_final_kernel<<<(N_NODES + 7) / 8, 256, 0, stream>>>(z1F, cls_w2, cls_b2, out, N_NODES);
}
